// Llama4MoE_60610578482062
// MI455X (gfx1250) — matrix operand order repaired and checked by reading, not compiled, not run on hardware
//
#include <hip/hip_runtime.h>
#include <hip/hip_bf16.h>

typedef __attribute__((ext_vector_type(16))) __bf16 v16bf;
typedef __attribute__((ext_vector_type(8)))  __bf16 v8bf;
typedef __attribute__((ext_vector_type(8)))  float  v8f;

#define T_  2048
#define D_  1024
#define F_  2048
#define FS_ 2048
#define E_  8

__device__ __forceinline__ float sigmoidf_(float x) {
  return __builtin_amdgcn_rcpf(1.0f + __expf(-x));
}
__device__ __forceinline__ float siluf_(float x) {
  return x * __builtin_amdgcn_rcpf(1.0f + __expf(-x));
}

__device__ __forceinline__ unsigned pack2bf(float a, float b) {
  union { __bf16 h[2]; unsigned u; } p;
  p.h[0] = (__bf16)a; p.h[1] = (__bf16)b;
  return p.u;
}

__device__ __forceinline__ v8f wmma_bf16(v16bf a, v16bf b, v8f c) {
  return __builtin_amdgcn_wmma_f32_16x16x32_bf16(
      /*neg_a=*/false, a, /*neg_b=*/false, b,
      /*c_mod=*/(short)0, c, /*reuse_a=*/false, /*reuse_b=*/false);
}

// A fragment: 16x32 bf16. lane<16: K=0..15 for row M=lane;
// lane>=16: K=16..31 for row M=lane-16.  lda = 32.
__device__ __forceinline__ v16bf load_frag_a(const __bf16* A, int lane) {
  int h = (lane >> 4) & 1, r = lane & 15;
  const __bf16* row = A + r * 32;
  v8bf lo = *(const v8bf*)(row + h * 16);
  v8bf hi = *(const v8bf*)(row + h * 16 + 8);
  v16bf a;
#pragma unroll
  for (int j = 0; j < 8; ++j) { a[j] = lo[j]; a[8 + j] = hi[j]; }
  return a;
}

// B fragment from LDS tile stored TRANSPOSED as Bt[N][K] (ldb=32):
// per lane 16 contiguous K starting at (lane>=16 ? 16 : 0), column N = lane&15.
__device__ __forceinline__ v16bf load_frag_bt(const __bf16* Bt, int lane) {
  int h = (lane >> 4) & 1, n = lane & 15;
  return *(const v16bf*)(Bt + n * 32 + h * 16);
}

// ---------------------------------------------------------------- router ----
__global__ void __launch_bounds__(256) zero_counts_kernel(int* counts) {
  if (threadIdx.x < E_) counts[threadIdx.x] = 0;
}

__global__ void __launch_bounds__(256)
router_kernel(const float* __restrict__ x, const float* __restrict__ rw,
              float* __restrict__ score, int* __restrict__ counts,
              int* __restrict__ tokens) {
  int wave = threadIdx.x >> 5;
  int lane = threadIdx.x & 31;
  int t = blockIdx.x * 8 + wave;
  if (t >= T_) return;
  float acc[E_];
#pragma unroll
  for (int e = 0; e < E_; ++e) acc[e] = 0.0f;
  for (int d = lane; d < D_; d += 32) {
    float xv = x[t * D_ + d];
#pragma unroll
    for (int e = 0; e < E_; ++e) acc[e] += xv * rw[e * D_ + d];
  }
#pragma unroll
  for (int e = 0; e < E_; ++e) {
#pragma unroll
    for (int off = 16; off > 0; off >>= 1) acc[e] += __shfl_xor(acc[e], off, 32);
  }
  if (lane == 0) {
    int best = 0; float bv = acc[0];
#pragma unroll
    for (int e = 1; e < E_; ++e) { if (acc[e] > bv) { bv = acc[e]; best = e; } }
    score[t] = sigmoidf_(bv);
    int pos = atomicAdd(&counts[best], 1);
    tokens[best * T_ + pos] = t;
  }
}

// -------------------------------------------------- shared expert GEMM 1 ----
// hS[t][f] = silu(x@ws_gate) * (x@ws_up), bf16.  Block tile 128x64, wave 32x32.
__global__ void __launch_bounds__(256)
shared_gemm1(const float* __restrict__ x, const float* __restrict__ wg,
             const float* __restrict__ wu, __bf16* __restrict__ hS) {
  __shared__ __align__(32) __bf16 lA[2][128 * 32];
  __shared__ __align__(32) __bf16 lBg[2][64 * 32];   // transposed [n][k]
  __shared__ __align__(32) __bf16 lBu[2][64 * 32];
  const int f0 = blockIdx.x * 64, t0 = blockIdx.y * 128;
  const int tid = threadIdx.x, lane = tid & 31, w = tid >> 5;
  const int wm = w >> 1, wn = w & 1;

  // staging coords: A = 128x32 (16 floats/thread), B = 32x64 (2k x 4n /thread)
  const int ar = tid >> 1, akc = (tid & 1) * 16;
  const int bk = (tid & 15) * 2, bn = (tid >> 4) * 4;

  float pa[16], pg0[4], pg1[4], pu0[4], pu1[4];

  auto load_regs = [&](int k0) {
    const float* sa = x + (size_t)(t0 + ar) * D_ + k0 + akc;
#pragma unroll
    for (int j = 0; j < 16; ++j) pa[j] = sa[j];
    const float* sg0 = wg + (size_t)(k0 + bk) * FS_ + f0 + bn;
    const float* su0 = wu + (size_t)(k0 + bk) * FS_ + f0 + bn;
#pragma unroll
    for (int j = 0; j < 4; ++j) {
      pg0[j] = sg0[j]; pg1[j] = sg0[FS_ + j];
      pu0[j] = su0[j]; pu1[j] = su0[FS_ + j];
    }
  };
  auto store_lds = [&](int buf) {
    v8bf t0v, t1v;
#pragma unroll
    for (int j = 0; j < 8; ++j) { t0v[j] = (__bf16)pa[j]; t1v[j] = (__bf16)pa[8 + j]; }
    __bf16* da = &lA[buf][ar * 32 + akc];
    *(v8bf*)da = t0v; *(v8bf*)(da + 8) = t1v;
#pragma unroll
    for (int j = 0; j < 4; ++j) {
      *(unsigned*)&lBg[buf][(bn + j) * 32 + bk] = pack2bf(pg0[j], pg1[j]);
      *(unsigned*)&lBu[buf][(bn + j) * 32 + bk] = pack2bf(pu0[j], pu1[j]);
    }
  };

  v8f cg[2][2] = {}; v8f cu[2][2] = {};
  load_regs(0); store_lds(0); __syncthreads();
  const int NIT = D_ / 32;
  for (int it = 0; it < NIT; ++it) {
    int cur = it & 1;
    if (it + 1 < NIT) load_regs((it + 1) * 32);
    const __bf16* A = lA[cur]; const __bf16* Bg = lBg[cur]; const __bf16* Bu = lBu[cur];
    v16bf a0  = load_frag_a(A + (32 * wm) * 32, lane);
    v16bf a1  = load_frag_a(A + (32 * wm + 16) * 32, lane);
    v16bf bg0 = load_frag_bt(Bg + (32 * wn) * 32, lane);
    v16bf bg1 = load_frag_bt(Bg + (32 * wn + 16) * 32, lane);
    v16bf bu0 = load_frag_bt(Bu + (32 * wn) * 32, lane);
    v16bf bu1 = load_frag_bt(Bu + (32 * wn + 16) * 32, lane);
    cg[0][0] = wmma_bf16(a0, bg0, cg[0][0]);
    cg[0][1] = wmma_bf16(a0, bg1, cg[0][1]);
    cg[1][0] = wmma_bf16(a1, bg0, cg[1][0]);
    cg[1][1] = wmma_bf16(a1, bg1, cg[1][1]);
    cu[0][0] = wmma_bf16(a0, bu0, cu[0][0]);
    cu[0][1] = wmma_bf16(a0, bu1, cu[0][1]);
    cu[1][0] = wmma_bf16(a1, bu0, cu[1][0]);
    cu[1][1] = wmma_bf16(a1, bu1, cu[1][1]);
    if (it + 1 < NIT) store_lds(cur ^ 1);
    __syncthreads();
  }
  int h = lane >> 4, n = lane & 15;
#pragma unroll
  for (int mi = 0; mi < 2; ++mi) {
#pragma unroll
    for (int i = 0; i < 8; ++i) {
      int m = t0 + 32 * wm + 16 * mi + i + 8 * h;
#pragma unroll
      for (int ni = 0; ni < 2; ++ni) {
        int c = f0 + 32 * wn + 16 * ni + n;
        hS[(size_t)m * FS_ + c] = (__bf16)(siluf_(cg[mi][ni][i]) * cu[mi][ni][i]);
      }
    }
  }
}

// -------------------------------------------------- shared expert GEMM 2 ----
// out[t][d] = hS @ ws_down  (pure store; routed path accumulates afterwards)
__global__ void __launch_bounds__(256)
shared_gemm2(const __bf16* __restrict__ hS, const float* __restrict__ wd,
             float* __restrict__ out) {
  __shared__ __align__(32) __bf16 lA[2][128 * 32];
  __shared__ __align__(32) __bf16 lB[2][64 * 32];
  const int d0 = blockIdx.x * 64, t0 = blockIdx.y * 128;
  const int tid = threadIdx.x, lane = tid & 31, w = tid >> 5;
  const int wm = w >> 1, wn = w & 1;
  const int ar = tid >> 1, akc = (tid & 1) * 16;
  const int bk = (tid & 15) * 2, bn = (tid >> 4) * 4;

  v8bf pa0, pa1; float pb0[4], pb1[4];
  auto load_regs = [&](int k0) {
    const __bf16* sa = hS + (size_t)(t0 + ar) * FS_ + k0 + akc;
    pa0 = *(const v8bf*)sa; pa1 = *(const v8bf*)(sa + 8);
    const float* sb = wd + (size_t)(k0 + bk) * D_ + d0 + bn;
#pragma unroll
    for (int j = 0; j < 4; ++j) { pb0[j] = sb[j]; pb1[j] = sb[D_ + j]; }
  };
  auto store_lds = [&](int buf) {
    __bf16* da = &lA[buf][ar * 32 + akc];
    *(v8bf*)da = pa0; *(v8bf*)(da + 8) = pa1;
#pragma unroll
    for (int j = 0; j < 4; ++j)
      *(unsigned*)&lB[buf][(bn + j) * 32 + bk] = pack2bf(pb0[j], pb1[j]);
  };

  v8f c[2][2] = {};
  load_regs(0); store_lds(0); __syncthreads();
  const int NIT = FS_ / 32;
  for (int it = 0; it < NIT; ++it) {
    int cur = it & 1;
    if (it + 1 < NIT) load_regs((it + 1) * 32);
    const __bf16* A = lA[cur]; const __bf16* B = lB[cur];
    v16bf a0 = load_frag_a(A + (32 * wm) * 32, lane);
    v16bf a1 = load_frag_a(A + (32 * wm + 16) * 32, lane);
    v16bf b0 = load_frag_bt(B + (32 * wn) * 32, lane);
    v16bf b1 = load_frag_bt(B + (32 * wn + 16) * 32, lane);
    c[0][0] = wmma_bf16(a0, b0, c[0][0]);
    c[0][1] = wmma_bf16(a0, b1, c[0][1]);
    c[1][0] = wmma_bf16(a1, b0, c[1][0]);
    c[1][1] = wmma_bf16(a1, b1, c[1][1]);
    if (it + 1 < NIT) store_lds(cur ^ 1);
    __syncthreads();
  }
  int h = lane >> 4, n = lane & 15;
#pragma unroll
  for (int mi = 0; mi < 2; ++mi) {
#pragma unroll
    for (int i = 0; i < 8; ++i) {
      int m = t0 + 32 * wm + 16 * mi + i + 8 * h;
#pragma unroll
      for (int ni = 0; ni < 2; ++ni) {
        out[(size_t)m * D_ + d0 + 32 * wn + 16 * ni + n] = c[mi][ni][i];
      }
    }
  }
}

// --------------------------------------------------- routed expert GEMM 1 ---
// hR[tok][f] = silu((s*x)@wg_e) * ((s*x)@wu_e), rows gathered via token list
__global__ void __launch_bounds__(256)
routed_gemm1(const float* __restrict__ x, const float* __restrict__ wg,
             const float* __restrict__ wu, const int* __restrict__ counts,
             const int* __restrict__ tokens, const float* __restrict__ score,
             __bf16* __restrict__ hR) {
  const int e = blockIdx.z;
  const int n_e = counts[e];
  const int t0 = blockIdx.y * 128;
  if (t0 >= n_e) return;  // uniform early exit, before any barrier
  const int f0 = blockIdx.x * 64;
  const int* tl = tokens + e * T_;
  const float* wge = wg + (size_t)e * D_ * F_;
  const float* wue = wu + (size_t)e * D_ * F_;
  __shared__ __align__(32) __bf16 lA[2][128 * 32];
  __shared__ __align__(32) __bf16 lBg[2][64 * 32];
  __shared__ __align__(32) __bf16 lBu[2][64 * 32];
  const int tid = threadIdx.x, lane = tid & 31, w = tid >> 5;
  const int wm = w >> 1, wn = w & 1;
  const int ar = tid >> 1, akc = (tid & 1) * 16;
  const int bk = (tid & 15) * 2, bn = (tid >> 4) * 4;

  // per-thread gather row (constant across K loop)
  const int am = t0 + ar;
  const int atok = (am < n_e) ? tl[am] : -1;
  const float ascale = (atok >= 0) ? score[atok] : 0.0f;
  const float* abase = x + (size_t)(atok >= 0 ? atok : 0) * D_;

  float pa[16], pg0[4], pg1[4], pu0[4], pu1[4];
  auto load_regs = [&](int k0) {
    const float* sa = abase + k0 + akc;
#pragma unroll
    for (int j = 0; j < 16; ++j) pa[j] = sa[j];
    const float* sg0 = wge + (size_t)(k0 + bk) * F_ + f0 + bn;
    const float* su0 = wue + (size_t)(k0 + bk) * F_ + f0 + bn;
#pragma unroll
    for (int j = 0; j < 4; ++j) {
      pg0[j] = sg0[j]; pg1[j] = sg0[F_ + j];
      pu0[j] = su0[j]; pu1[j] = su0[F_ + j];
    }
  };
  auto store_lds = [&](int buf) {
    v8bf t0v, t1v;
#pragma unroll
    for (int j = 0; j < 8; ++j) {
      t0v[j] = (__bf16)(ascale * pa[j]);        // scale==0 zeroes padded rows
      t1v[j] = (__bf16)(ascale * pa[8 + j]);
    }
    __bf16* da = &lA[buf][ar * 32 + akc];
    *(v8bf*)da = t0v; *(v8bf*)(da + 8) = t1v;
#pragma unroll
    for (int j = 0; j < 4; ++j) {
      *(unsigned*)&lBg[buf][(bn + j) * 32 + bk] = pack2bf(pg0[j], pg1[j]);
      *(unsigned*)&lBu[buf][(bn + j) * 32 + bk] = pack2bf(pu0[j], pu1[j]);
    }
  };

  v8f cg[2][2] = {}; v8f cu[2][2] = {};
  load_regs(0); store_lds(0); __syncthreads();
  const int NIT = D_ / 32;
  for (int it = 0; it < NIT; ++it) {
    int cur = it & 1;
    if (it + 1 < NIT) load_regs((it + 1) * 32);
    const __bf16* A = lA[cur]; const __bf16* Bg = lBg[cur]; const __bf16* Bu = lBu[cur];
    v16bf a0  = load_frag_a(A + (32 * wm) * 32, lane);
    v16bf a1  = load_frag_a(A + (32 * wm + 16) * 32, lane);
    v16bf bg0 = load_frag_bt(Bg + (32 * wn) * 32, lane);
    v16bf bg1 = load_frag_bt(Bg + (32 * wn + 16) * 32, lane);
    v16bf bu0 = load_frag_bt(Bu + (32 * wn) * 32, lane);
    v16bf bu1 = load_frag_bt(Bu + (32 * wn + 16) * 32, lane);
    cg[0][0] = wmma_bf16(a0, bg0, cg[0][0]);
    cg[0][1] = wmma_bf16(a0, bg1, cg[0][1]);
    cg[1][0] = wmma_bf16(a1, bg0, cg[1][0]);
    cg[1][1] = wmma_bf16(a1, bg1, cg[1][1]);
    cu[0][0] = wmma_bf16(a0, bu0, cu[0][0]);
    cu[0][1] = wmma_bf16(a0, bu1, cu[0][1]);
    cu[1][0] = wmma_bf16(a1, bu0, cu[1][0]);
    cu[1][1] = wmma_bf16(a1, bu1, cu[1][1]);
    if (it + 1 < NIT) store_lds(cur ^ 1);
    __syncthreads();
  }
  int h = lane >> 4, n = lane & 15;
#pragma unroll
  for (int mi = 0; mi < 2; ++mi) {
#pragma unroll
    for (int i = 0; i < 8; ++i) {
      int m = t0 + 32 * wm + 16 * mi + i + 8 * h;
      if (m < n_e) {
        int tok = tl[m];
#pragma unroll
        for (int ni = 0; ni < 2; ++ni) {
          int c = f0 + 32 * wn + 16 * ni + n;
          hR[(size_t)tok * F_ + c] = (__bf16)(siluf_(cg[mi][ni][i]) * cu[mi][ni][i]);
        }
      }
    }
  }
}

// --------------------------------------------------- routed down-proj -------
// out[tok][d] += hR[tok] @ w_down[e]   (disjoint tokens: no atomics needed)
__global__ void __launch_bounds__(256)
routed_down(const __bf16* __restrict__ hR, const float* __restrict__ wdn,
            const int* __restrict__ counts, const int* __restrict__ tokens,
            float* __restrict__ out) {
  const int e = blockIdx.z;
  const int n_e = counts[e];
  const int t0 = blockIdx.y * 128;
  if (t0 >= n_e) return;
  const int d0 = blockIdx.x * 64;
  const int* tl = tokens + e * T_;
  const float* wde = wdn + (size_t)e * F_ * D_;
  __shared__ __align__(32) __bf16 lA[2][128 * 32];
  __shared__ __align__(32) __bf16 lB[2][64 * 32];
  const int tid = threadIdx.x, lane = tid & 31, w = tid >> 5;
  const int wm = w >> 1, wn = w & 1;
  const int ar = tid >> 1, akc = (tid & 1) * 16;
  const int bk = (tid & 15) * 2, bn = (tid >> 4) * 4;

  const int am = t0 + ar;
  const int atok = (am < n_e) ? tl[am] : -1;
  const __bf16* abase = hR + (size_t)(atok >= 0 ? atok : 0) * F_;

  v8bf pa0, pa1; float pb0[4], pb1[4];
  auto load_regs = [&](int k0) {
    const __bf16* sa = abase + k0 + akc;
    pa0 = *(const v8bf*)sa; pa1 = *(const v8bf*)(sa + 8);
    const float* sb = wde + (size_t)(k0 + bk) * D_ + d0 + bn;
#pragma unroll
    for (int j = 0; j < 4; ++j) { pb0[j] = sb[j]; pb1[j] = sb[D_ + j]; }
  };
  auto store_lds = [&](int buf) {
    __bf16* da = &lA[buf][ar * 32 + akc];
    if (atok >= 0) {
      *(v8bf*)da = pa0; *(v8bf*)(da + 8) = pa1;
    } else {
      v8bf z = {};
      *(v8bf*)da = z; *(v8bf*)(da + 8) = z;
    }
#pragma unroll
    for (int j = 0; j < 4; ++j)
      *(unsigned*)&lB[buf][(bn + j) * 32 + bk] = pack2bf(pb0[j], pb1[j]);
  };

  v8f c[2][2] = {};
  load_regs(0); store_lds(0); __syncthreads();
  const int NIT = F_ / 32;
  for (int it = 0; it < NIT; ++it) {
    int cur = it & 1;
    if (it + 1 < NIT) load_regs((it + 1) * 32);
    const __bf16* A = lA[cur]; const __bf16* B = lB[cur];
    v16bf a0 = load_frag_a(A + (32 * wm) * 32, lane);
    v16bf a1 = load_frag_a(A + (32 * wm + 16) * 32, lane);
    v16bf b0 = load_frag_bt(B + (32 * wn) * 32, lane);
    v16bf b1 = load_frag_bt(B + (32 * wn + 16) * 32, lane);
    c[0][0] = wmma_bf16(a0, b0, c[0][0]);
    c[0][1] = wmma_bf16(a0, b1, c[0][1]);
    c[1][0] = wmma_bf16(a1, b0, c[1][0]);
    c[1][1] = wmma_bf16(a1, b1, c[1][1]);
    if (it + 1 < NIT) store_lds(cur ^ 1);
    __syncthreads();
  }
  int h = lane >> 4, n = lane & 15;
#pragma unroll
  for (int mi = 0; mi < 2; ++mi) {
#pragma unroll
    for (int i = 0; i < 8; ++i) {
      int m = t0 + 32 * wm + 16 * mi + i + 8 * h;
      if (m < n_e) {
        int tok = tl[m];
#pragma unroll
        for (int ni = 0; ni < 2; ++ni) {
          int cc = d0 + 32 * wn + 16 * ni + n;
          out[(size_t)tok * D_ + cc] += c[mi][ni][i];
        }
      }
    }
  }
}

// ---------------------------------------------------------------- launch ----
extern "C" void kernel_launch(void* const* d_in, const int* in_sizes, int n_in,
                              void* d_out, int out_size, void* d_ws, size_t ws_size,
                              hipStream_t stream) {
  (void)in_sizes; (void)n_in; (void)out_size; (void)ws_size;
  const float* x   = (const float*)d_in[0];  // [T, D]
  const float* rw  = (const float*)d_in[1];  // [E, D]
  const float* wg  = (const float*)d_in[2];  // [E, D, F]
  const float* wu  = (const float*)d_in[3];  // [E, D, F]
  const float* wd  = (const float*)d_in[4];  // [E, F, D]
  const float* wsg = (const float*)d_in[5];  // [D, FS]
  const float* wsu = (const float*)d_in[6];  // [D, FS]
  const float* wsd = (const float*)d_in[7];  // [FS, D]
  float* out = (float*)d_out;                // [T, D] fp32

  char* ws = (char*)d_ws;
  size_t off = 0;
  int*    counts = (int*)(ws + off);    off += 256;
  float*  score  = (float*)(ws + off);  off += (size_t)T_ * 4;
  int*    tokens = (int*)(ws + off);    off += (size_t)E_ * T_ * 4;
  __bf16* hS     = (__bf16*)(ws + off); off += (size_t)T_ * FS_ * 2;
  __bf16* hR     = (__bf16*)(ws + off); off += (size_t)T_ * F_ * 2;

  zero_counts_kernel<<<1, 256, 0, stream>>>(counts);
  router_kernel<<<T_ / 8, 256, 0, stream>>>(x, rw, score, counts, tokens);
  shared_gemm1<<<dim3(FS_ / 64, T_ / 128), 256, 0, stream>>>(x, wsg, wsu, hS);
  shared_gemm2<<<dim3(D_ / 64, T_ / 128), 256, 0, stream>>>(hS, wsd, out);
  routed_gemm1<<<dim3(F_ / 64, T_ / 128, E_), 256, 0, stream>>>(x, wg, wu, counts, tokens, score, hR);
  routed_down<<<dim3(D_ / 64, T_ / 128, E_), 256, 0, stream>>>(hR, wd, counts, tokens, out);
}
